// NaViT_66529043415477
// MI455X (gfx1250) — compile-verified, matrix operand order checked by reading, not run on hardware
//
#include <hip/hip_runtime.h>
#include <hip/hip_bf16.h>
#include <math.h>
#include <stdint.h>

typedef __attribute__((ext_vector_type(16))) _Float16 v16h;
typedef __attribute__((ext_vector_type(8)))  float    v8f;

#define HEADS 12
#define NTOK  1024
#define DH    64
#define DIMM  768

__device__ __forceinline__ v8f vzero8() {
    v8f z;
#pragma unroll
    for (int i = 0; i < 8; i++) z[i] = 0.0f;
    return z;
}

__device__ __forceinline__ v8f wmma16(v16h a, v16h b, v8f c) {
    return __builtin_amdgcn_wmma_f32_16x16x32_f16(false, a, false, b, (short)0, c, false, false);
}

// LDS offset of a shared-memory pointer (generic LDS addr keeps offset in [31:0])
__device__ __forceinline__ uint32_t lds_off_u32(const void* p) {
    return (uint32_t)(uintptr_t)p;
}

// CDNA5 async DMA: 16B per lane, global -> LDS, tracked by ASYNCcnt
__device__ __forceinline__ void async_b128(uint32_t lds_byte_off, const void* gptr) {
    asm volatile("global_load_async_to_lds_b128 %0, %1, off"
                 :: "v"(lds_byte_off), "v"(gptr) : "memory");
}
__device__ __forceinline__ void wait_async0()  { asm volatile("s_wait_asynccnt 0"  ::: "memory"); }
__device__ __forceinline__ void wait_async16() { asm volatile("s_wait_asynccnt 16" ::: "memory"); }

// Copy a 64x64-half tile (row stride `srcStride` halves) into LDS [64][72] via async DMA.
// 512 16B-chunks, 128 threads -> 4 per thread (4 async instructions per wave).
__device__ __forceinline__ void copy_tile64_async(_Float16* dst, const _Float16* src,
                                                  size_t srcStride, int tid) {
#pragma unroll
    for (int c = 0; c < 4; c++) {
        int ch = tid + c * 128;
        int r = ch >> 3, cc = ch & 7;
        async_b128(lds_off_u32(dst + r * 72 + cc * 8),
                   (const void*)(src + (size_t)r * srcStride + cc * 8));
    }
}

// Copy a 128x64-half tile into LDS [128][72]: 1024 chunks, 128 threads -> 8 per thread.
__device__ __forceinline__ void copy_tile128_async(_Float16* dst, const _Float16* src,
                                                   size_t srcStride, int tid) {
#pragma unroll
    for (int c = 0; c < 8; c++) {
        int ch = tid + c * 128;
        int r = ch >> 3, cc = ch & 7;
        async_b128(lds_off_u32(dst + r * 72 + cc * 8),
                   (const void*)(src + (size_t)r * srcStride + cc * 8));
    }
}

// Load a 16x32 WMMA fragment (A layout; also used for B stored transposed as [n][k]).
// ISA 7.12.2: lanes 0-15 row=lane, K 0..7 & 16..23 ; lanes 16-31 row=lane-16, K 8..15 & 24..31.
__device__ __forceinline__ v16h ld_frag(const _Float16* __restrict__ base, int stride, int kb) {
    int lane = threadIdx.x & 31;
    int r    = lane & 15;
    int hi   = lane >> 4;
    const _Float16* p = base + r * stride + kb + hi * 8;
    union { v16h v; float4 f[2]; } u;
    u.f[0] = *(const float4*)(p);
    u.f[1] = *(const float4*)(p + 16);
    return u.v;
}

// ---------------------------------------------------------------------------
// GEMM: C[M,N] = epilogue(A[M,K] @ Bt[N,K]^T); A,Bt f16, f32 acc, WMMA.
// K multiple of 64; A rows padded to mult of 128; Bt rows padded to mult of 128.
// act: 0 none, 1 exact GELU. res_mode: 0 none, 1 full f32 matrix, 2 f32 row-vec.
// out_f16: 1 -> Cout is _Float16*, else float*.
// Block 128 threads (4 waves, 2x2), tile 128x128, wave tile 64x64,
// double-buffered async-DMA LDS stages of K=64.
// ---------------------------------------------------------------------------
__global__ __launch_bounds__(128)
void gemm_kernel(const _Float16* __restrict__ A, const _Float16* __restrict__ Bt,
                 const float* __restrict__ bias, const float* __restrict__ res,
                 void* __restrict__ Cout, int M, int N, int K,
                 int act, int res_mode, int out_f16)
{
    extern __shared__ char gsm[];
    _Float16* As0 = (_Float16*)gsm;        // 128*72
    _Float16* As1 = As0 + 128 * 72;
    _Float16* Bs0 = As1 + 128 * 72;
    _Float16* Bs1 = Bs0 + 128 * 72;

    int tid  = threadIdx.x;
    int lane = tid & 31;
    int wave = tid >> 5;
    int wm = (wave >> 1) * 64, wn = (wave & 1) * 64;
    int m0 = blockIdx.y * 128, n0 = blockIdx.x * 128;

    const _Float16* Abase = A  + (size_t)m0 * K;
    const _Float16* Bbase = Bt + (size_t)n0 * K;

    v8f acc[4][4];
#pragma unroll
    for (int i = 0; i < 4; i++)
#pragma unroll
        for (int j = 0; j < 4; j++) acc[i][j] = vzero8();

    int nk = K >> 6;  // stages of 64
    copy_tile128_async(As0, Abase, K, tid);
    copy_tile128_async(Bs0, Bbase, K, tid);

    for (int ks = 0; ks < nk; ks++) {
        int cur = ks & 1;
        _Float16* Ac = cur ? As1 : As0;
        _Float16* Bc = cur ? Bs1 : Bs0;
        if (ks + 1 < nk) {
            __syncthreads();  // everyone done computing from the other buffer
            _Float16* An = cur ? As0 : As1;
            _Float16* Bn = cur ? Bs0 : Bs1;
            copy_tile128_async(An, Abase + (size_t)(ks + 1) * 64, K, tid);
            copy_tile128_async(Bn, Bbase + (size_t)(ks + 1) * 64, K, tid);
            wait_async16();   // current stage's 16 copies complete (in-order)
        } else {
            wait_async0();
        }
        __syncthreads();

#pragma unroll
        for (int kb = 0; kb < 64; kb += 32) {
            v16h af[4], bf[4];
#pragma unroll
            for (int t = 0; t < 4; t++) af[t] = ld_frag(Ac + (wm + t * 16) * 72, 72, kb);
#pragma unroll
            for (int t = 0; t < 4; t++) bf[t] = ld_frag(Bc + (wn + t * 16) * 72, 72, kb);
#pragma unroll
            for (int ti = 0; ti < 4; ti++)
#pragma unroll
                for (int tj = 0; tj < 4; tj++)
                    acc[ti][tj] = wmma16(af[ti], bf[tj], acc[ti][tj]);
        }
    }

    int hi = lane >> 4, lc = lane & 15;
#pragma unroll
    for (int ti = 0; ti < 4; ti++)
#pragma unroll
        for (int tj = 0; tj < 4; tj++) {
            int col = n0 + wn + tj * 16 + lc;
#pragma unroll
            for (int r = 0; r < 8; r++) {
                int row = m0 + wm + ti * 16 + r + 8 * hi;
                if (row < M && col < N) {
                    float v = acc[ti][tj][r];
                    if (bias) v += bias[col];
                    if (act == 1) v = 0.5f * v * (1.0f + erff(v * 0.70710678118654752f));
                    if (res_mode == 1)      v += res[(size_t)row * N + col];
                    else if (res_mode == 2) v += res[col];
                    if (out_f16) ((_Float16*)Cout)[(size_t)row * N + col] = (_Float16)v;
                    else         ((float*)Cout)[(size_t)row * N + col] = v;
                }
            }
        }
}

// ---------------------------------------------------------------------------
// Tiled transpose + f32->f16 convert:  Wt[n][k] = (f16) W[k][n], n zero-padded to Npad.
// Block (32,8), grid (ceil(K/32), ceil(Npad/32)).
// ---------------------------------------------------------------------------
__global__ void wtrans_kernel(const float* __restrict__ W, _Float16* __restrict__ Wt,
                              int K, int N, int Npad)
{
    __shared__ float t[32][33];
    int k0 = blockIdx.x * 32, n0 = blockIdx.y * 32;
    int tx = threadIdx.x, ty = threadIdx.y;
    for (int i = ty; i < 32; i += 8) {
        int k = k0 + i, n = n0 + tx;
        t[i][tx] = (k < K && n < N) ? W[(size_t)k * N + n] : 0.0f;
    }
    __syncthreads();
    for (int i = ty; i < 32; i += 8) {
        int n = n0 + i, k = k0 + tx;
        if (n < Npad && k < K) Wt[(size_t)n * K + k] = (_Float16)t[tx][i];
    }
}

// ---------------------------------------------------------------------------
// LayerNorm (f32 in -> f16 out): one block per row.
// ---------------------------------------------------------------------------
__global__ void ln_kernel(const float* __restrict__ x, const float* __restrict__ g,
                          _Float16* __restrict__ y, int D)
{
    __shared__ float red[256];
    int row = blockIdx.x, tid = threadIdx.x;
    const float* xr = x + (size_t)row * D;
    _Float16* yr = y + (size_t)row * D;

    float s = 0.0f;
    for (int i = tid; i < D; i += 256) s += xr[i];
    red[tid] = s; __syncthreads();
    for (int st = 128; st > 0; st >>= 1) { if (tid < st) red[tid] += red[tid + st]; __syncthreads(); }
    float mean = red[0] / (float)D; __syncthreads();

    float s2 = 0.0f;
    for (int i = tid; i < D; i += 256) { float d = xr[i] - mean; s2 += d * d; }
    red[tid] = s2; __syncthreads();
    for (int st = 128; st > 0; st >>= 1) { if (tid < st) red[tid] += red[tid + st]; __syncthreads(); }
    float rstd = rsqrtf(red[0] / (float)D + 1e-5f);

    for (int i = tid; i < D; i += 256) yr[i] = (_Float16)((xr[i] - mean) * rstd * g[i]);
}

// LN (f16 in) + factorized 2D pos-embed add -> f32 residual stream
__global__ void embpos_kernel(const _Float16* __restrict__ x, const float* __restrict__ g,
                              const float* __restrict__ pos_h, const float* __restrict__ pos_w,
                              const int* __restrict__ pos_hw, float* __restrict__ y)
{
    __shared__ float red[256];
    int row = blockIdx.x, tid = threadIdx.x;
    const _Float16* xr = x + (size_t)row * DIMM;
    float* yr = y + (size_t)row * DIMM;

    float s = 0.0f;
    for (int i = tid; i < DIMM; i += 256) s += (float)xr[i];
    red[tid] = s; __syncthreads();
    for (int st = 128; st > 0; st >>= 1) { if (tid < st) red[tid] += red[tid + st]; __syncthreads(); }
    float mean = red[0] / (float)DIMM; __syncthreads();

    float s2 = 0.0f;
    for (int i = tid; i < DIMM; i += 256) { float d = (float)xr[i] - mean; s2 += d * d; }
    red[tid] = s2; __syncthreads();
    for (int st = 128; st > 0; st >>= 1) { if (tid < st) red[tid] += red[tid + st]; __syncthreads(); }
    float rstd = rsqrtf(red[0] / (float)DIMM + 1e-5f);

    int ph = pos_hw[row * 2 + 0];
    int pw = pos_hw[row * 2 + 1];
    for (int i = tid; i < DIMM; i += 256)
        yr[i] = ((float)xr[i] - mean) * rstd * g[i]
              + pos_h[(size_t)ph * DIMM + i] + pos_w[(size_t)pw * DIMM + i];
}

// Broadcast LN of one 768-vector into 16 identical f16 rows (pool query)
__global__ void bcast_ln_kernel(const float* __restrict__ v, const float* __restrict__ g,
                                _Float16* __restrict__ y)
{
    __shared__ float red[256];
    int tid = threadIdx.x;
    float s = 0.0f;
    for (int i = tid; i < DIMM; i += 256) s += v[i];
    red[tid] = s; __syncthreads();
    for (int st = 128; st > 0; st >>= 1) { if (tid < st) red[tid] += red[tid + st]; __syncthreads(); }
    float mean = red[0] / (float)DIMM; __syncthreads();
    float s2 = 0.0f;
    for (int i = tid; i < DIMM; i += 256) { float d = v[i] - mean; s2 += d * d; }
    red[tid] = s2; __syncthreads();
    for (int st = 128; st > 0; st >>= 1) { if (tid < st) red[tid] += red[tid + st]; __syncthreads(); }
    float rstd = rsqrtf(red[0] / (float)DIMM + 1e-5f);
    for (int i = tid; i < DIMM; i += 256) {
        _Float16 o = (_Float16)((v[i] - mean) * rstd * g[i]);
        for (int r = 0; r < 16; r++) y[(size_t)r * DIMM + i] = o;
    }
}

// ---------------------------------------------------------------------------
// Reorder (B,Ntok,INNER)-f16 -> f16 head-split, with optional RMS-QK norm.
// mode 0: copy -> (B,H,Ntok,DH)
// mode 1: rms  -> (B,H,Ntok,DH):  y = x / max(||x||,1e-12) * 8 * gamma[h,d]
// mode 2: copy -> (B,H,DH,Ntok)   (transposed, for V)
// grid (H, Ntok, B), block 64
// ---------------------------------------------------------------------------
__global__ void qk_reorder_kernel(const _Float16* __restrict__ src, int srcStride, int srcOff,
                                  const float* __restrict__ gamma, _Float16* __restrict__ dst,
                                  int Ntok, int mode)
{
    __shared__ float red[64];
    int h = blockIdx.x, n = blockIdx.y, b = blockIdx.z, d = threadIdx.x;
    float v = (float)src[(size_t)(b * Ntok + n) * srcStride + srcOff + h * DH + d];
    if (mode == 1) {
        red[d] = v * v; __syncthreads();
        for (int st = 32; st > 0; st >>= 1) { if (d < st) red[d] += red[d + st]; __syncthreads(); }
        float nrm = sqrtf(red[0]);
        v = v / fmaxf(nrm, 1e-12f) * 8.0f * gamma[h * DH + d];
    }
    if (mode == 2)
        dst[(((size_t)b * HEADS + h) * DH + d) * Ntok + n] = (_Float16)v;
    else
        dst[(((size_t)b * HEADS + h) * Ntok + n) * DH + d] = (_Float16)v;
}

// ---------------------------------------------------------------------------
// Fused masked attention per (row-block 64, head, batch). Block: 128 threads.
// qh,kh: (B,H,N,DH) f16; vt: (B,H,DH,N) f16 (pre-transposed).
// Dynamic LDS: Qs[64*72], KVs[64*72], P[64*1024] f16, ids[1024], pad[1024]
// ---------------------------------------------------------------------------
__global__ __launch_bounds__(128)
void attn_kernel(const _Float16* __restrict__ qh, const _Float16* __restrict__ kh,
                 const _Float16* __restrict__ vt, const int* __restrict__ ids,
                 const unsigned char* __restrict__ pad, _Float16* __restrict__ out)
{
    extern __shared__ char smem[];
    _Float16* Qs  = (_Float16*)smem;                  // 64*72
    _Float16* KVs = Qs + 64 * 72;                     // 64*72
    _Float16* P   = KVs + 64 * 72;                    // 64*1024
    int* sid          = (int*)(P + 64 * 1024);        // 1024
    unsigned char* sp = (unsigned char*)(sid + 1024); // 1024

    int ib = blockIdx.x, h = blockIdx.y, b = blockIdx.z;
    int tid = threadIdx.x, lane = tid & 31, wave = tid >> 5;
    int i0 = ib * 64;

    const _Float16* qb = qh + (((size_t)b * HEADS + h) * NTOK) * DH;
    const _Float16* kb = kh + (((size_t)b * HEADS + h) * NTOK) * DH;
    const _Float16* vb = vt + (((size_t)b * HEADS + h) * DH) * NTOK;

    copy_tile64_async(Qs, qb + (size_t)i0 * DH, DH, tid);
    for (int i = tid; i < NTOK; i += 128) {
        sid[i] = ids[b * NTOK + i];
        sp[i]  = pad[b * NTOK + i];
    }
    wait_async0();

    // ---- pass 1: S = Q @ K^T  (raw scores into P, f16) ----
    for (int j = 0; j < 16; j++) {
        __syncthreads();
        copy_tile64_async(KVs, kb + (size_t)(j * 64) * DH, DH, tid);
        wait_async0();
        __syncthreads();
        v8f acc[4];
#pragma unroll
        for (int t = 0; t < 4; t++) acc[t] = vzero8();
#pragma unroll
        for (int kk = 0; kk < 64; kk += 32) {
            v16h a = ld_frag(Qs + (wave * 16) * 72, 72, kk);
#pragma unroll
            for (int t = 0; t < 4; t++) {
                v16h bf = ld_frag(KVs + (t * 16) * 72, 72, kk);
                acc[t] = wmma16(a, bf, acc[t]);
            }
        }
        int hi = lane >> 4, lc = lane & 15;
#pragma unroll
        for (int t = 0; t < 4; t++)
#pragma unroll
            for (int r = 0; r < 8; r++) {
                int rr = wave * 16 + r + 8 * hi;
                P[rr * 1024 + j * 64 + t * 16 + lc] = (_Float16)acc[t][r];
            }
    }
    __syncthreads();

    // ---- pass 2: masked softmax, one thread per row ----
    if (tid < 64) {
        int r = tid;
        int myid = sid[i0 + r];
        float mx = -1e30f;
        for (int c = 0; c < NTOK; c++) {
            bool ok = (sid[c] == myid) && sp[c];
            float s = ok ? (float)P[r * 1024 + c] : -1e30f;
            mx = fmaxf(mx, s);
        }
        float sum = 0.0f;
        for (int c = 0; c < NTOK; c++) {
            bool ok = (sid[c] == myid) && sp[c];
            float s = ok ? (float)P[r * 1024 + c] : -1e30f;
            float e = __expf(s - mx);
            sum += e;
            P[r * 1024 + c] = (_Float16)e;
        }
        float inv = 1.0f / sum;
        for (int c = 0; c < NTOK; c++)
            P[r * 1024 + c] = (_Float16)((float)P[r * 1024 + c] * inv);
    }
    __syncthreads();

    // ---- pass 3: O = P @ V  (V tile loaded pre-transposed: [d][c]) ----
    v8f o[4];
#pragma unroll
    for (int t = 0; t < 4; t++) o[t] = vzero8();
    for (int j = 0; j < 16; j++) {
        __syncthreads();
        copy_tile64_async(KVs, vb + (size_t)(j * 64), NTOK, tid);
        wait_async0();
        __syncthreads();
#pragma unroll
        for (int kk = 0; kk < 64; kk += 32) {
            v16h a = ld_frag(P + (wave * 16) * 1024 + j * 64, 1024, kk);
#pragma unroll
            for (int t = 0; t < 4; t++) {
                v16h bf = ld_frag(KVs + (t * 16) * 72, 72, kk);
                o[t] = wmma16(a, bf, o[t]);
            }
        }
    }
    int hi = lane >> 4, lc = lane & 15;
#pragma unroll
    for (int t = 0; t < 4; t++)
#pragma unroll
        for (int r = 0; r < 8; r++) {
            int row = i0 + wave * 16 + r + 8 * hi;
            int d = t * 16 + lc;
            out[((size_t)b * NTOK + row) * (HEADS * DH) + h * DH + d] = (_Float16)o[t][r];
        }
}

// ---------------------------------------------------------------------------
// Attention pooling: 4 learned queries per image group (tiny) -> scalar kernel.
// vt is (B,H,DH,N). grid (N_IMG=4, H, B), block 128. out: f16 (128x768 padded).
// ---------------------------------------------------------------------------
__global__ void pool_attn_kernel(const _Float16* __restrict__ qph, const _Float16* __restrict__ kh,
                                 const _Float16* __restrict__ vt, const int* __restrict__ ids,
                                 const unsigned char* __restrict__ pad, _Float16* __restrict__ out)
{
    __shared__ float qv[64];
    __shared__ float probs[NTOK];
    __shared__ float red[128];
    int img = blockIdx.x, h = blockIdx.y, b = blockIdx.z, tid = threadIdx.x;

    if (tid < 64) qv[tid] = (float)qph[(((size_t)b * HEADS + h) * 4 + img) * DH + tid];
    __syncthreads();
    const _Float16* kb = kh + ((size_t)b * HEADS + h) * NTOK * DH;
    const _Float16* vb = vt + ((size_t)b * HEADS + h) * DH * NTOK;

    float lmax = -1e30f;
    for (int j = tid; j < NTOK; j += 128) {
        float s = 0.0f;
        for (int d = 0; d < 64; d++) s += qv[d] * (float)kb[(size_t)j * DH + d];
        bool ok = (ids[b * NTOK + j] == img) && pad[b * NTOK + j];
        s = ok ? s : -1e30f;
        probs[j] = s;
        lmax = fmaxf(lmax, s);
    }
    red[tid] = lmax; __syncthreads();
    for (int st = 64; st > 0; st >>= 1) { if (tid < st) red[tid] = fmaxf(red[tid], red[tid + st]); __syncthreads(); }
    float mx = red[0]; __syncthreads();

    float ls = 0.0f;
    for (int j = tid; j < NTOK; j += 128) { float e = __expf(probs[j] - mx); probs[j] = e; ls += e; }
    red[tid] = ls; __syncthreads();
    for (int st = 64; st > 0; st >>= 1) { if (tid < st) red[tid] += red[tid + st]; __syncthreads(); }
    float inv = 1.0f / red[0]; __syncthreads();

    if (tid < 64) {
        float acc = 0.0f;
        for (int j = 0; j < NTOK; j++) acc += probs[j] * (float)vb[(size_t)tid * NTOK + j];
        out[((size_t)b * 4 + img) * (HEADS * DH) + h * DH + tid] = (_Float16)(acc * inv);
    }
}

// ---------------------------------------------------------------------------
// Host side
// ---------------------------------------------------------------------------
static const size_t GEMM_SMEM = (size_t)4 * 128 * 72 * 2;  // 73728 B

static inline void launch_gemm(const _Float16* A, const _Float16* Bt, const float* bias,
                               const float* res, void* C, int M, int N, int K,
                               int act, int res_mode, int out_f16, hipStream_t s)
{
    dim3 g((N + 127) / 128, (M + 127) / 128);
    gemm_kernel<<<g, dim3(128), GEMM_SMEM, s>>>(A, Bt, bias, res, C, M, N, K, act, res_mode, out_f16);
}

static inline void launch_wtrans(const float* W, _Float16* Wt, int K, int N, int Npad,
                                 hipStream_t s)
{
    dim3 g((K + 31) / 32, (Npad + 31) / 32);
    wtrans_kernel<<<g, dim3(32, 8), 0, s>>>(W, Wt, K, N, Npad);
}

extern "C" void kernel_launch(void* const* d_in, const int* in_sizes, int n_in,
                              void* d_out, int out_size, void* d_ws, size_t ws_size,
                              hipStream_t stream)
{
    (void)in_sizes; (void)n_in; (void)out_size; (void)ws_size;
    const int B = 4, NI = 4;
    const int M = B * NTOK;              // 4096 rows
    const int INNER = HEADS * DH;        // 768
    const int MLPD = 3072;

    // pytree (sorted-dict-key) flattening of setup_inputs():
    const int*           image_ids = (const int*)d_in[0];
    const unsigned char* key_pad   = (const unsigned char*)d_in[1];
    const float* b_embed    = (const float*)d_in[2];
    const float* emb_ln_g   = (const float*)d_in[3];
    const float* final_ln_g = (const float*)d_in[4];
    const float* head_ln_g  = (const float*)d_in[5];
    struct Layer { const float *kn_g,*a_norm_g,*qn_g,*wkv,*wo,*wq,*b1,*b2,*f_norm_g,*w1,*w2; } L[2];
    int idx = 6;
    for (int l = 0; l < 2; l++) {
        L[l].kn_g     = (const float*)d_in[idx++];
        L[l].a_norm_g = (const float*)d_in[idx++];
        L[l].qn_g     = (const float*)d_in[idx++];
        L[l].wkv      = (const float*)d_in[idx++];
        L[l].wo       = (const float*)d_in[idx++];
        L[l].wq       = (const float*)d_in[idx++];
        L[l].b1       = (const float*)d_in[idx++];
        L[l].b2       = (const float*)d_in[idx++];
        L[l].f_norm_g = (const float*)d_in[idx++];
        L[l].w1       = (const float*)d_in[idx++];
        L[l].w2       = (const float*)d_in[idx++];
    }
    const float* pe_ln_g  = (const float*)d_in[idx++];   // 28
    const float* p_kn_g   = (const float*)d_in[idx++];
    const float* p_norm_g = (const float*)d_in[idx++];
    const float* p_qn_g   = (const float*)d_in[idx++];
    const float* p_wkv    = (const float*)d_in[idx++];
    const float* p_wo     = (const float*)d_in[idx++];
    const float* p_wq     = (const float*)d_in[idx++];
    const float* pool_q   = (const float*)d_in[idx++];   // 35
    const float* pos_h    = (const float*)d_in[idx++];
    const float* pos_w    = (const float*)d_in[idx++];
    const float* w_embed  = (const float*)d_in[idx++];
    const float* w_head   = (const float*)d_in[idx++];
    const float* patches  = (const float*)d_in[idx++];   // 40
    const int*   pos_hw   = (const int*)d_in[idx++];     // 41

    // workspace carve-up
    char* wsp = (char*)d_ws;
    size_t off = 0;
    auto take = [&](size_t bytes) -> void* {
        void* p = wsp + off;
        off += (bytes + 255) & ~(size_t)255;
        return p;
    };
    // activations
    _Float16* xn16   = (_Float16*)take((size_t)M * DIMM * 2);      // LN output (GEMM A)
    float*    x      = (float*)take((size_t)M * DIMM * 4);         // residual stream (f32)
    _Float16* aout16 = (_Float16*)take((size_t)M * DIMM * 2);      // embed out / attn out
    _Float16* q16    = (_Float16*)take((size_t)M * DIMM * 2);
    _Float16* kv16   = (_Float16*)take((size_t)M * 2 * INNER * 2);
    _Float16* h16    = (_Float16*)take((size_t)M * MLPD * 2);
    _Float16* qh     = (_Float16*)take((size_t)B * HEADS * NTOK * DH * 2);
    _Float16* kh     = (_Float16*)take((size_t)B * HEADS * NTOK * DH * 2);
    _Float16* vt     = (_Float16*)take((size_t)B * HEADS * DH * NTOK * 2);
    _Float16* qph    = (_Float16*)take((size_t)B * HEADS * NI * DH * 2);
    _Float16* pat16  = (_Float16*)take((size_t)128 * DIMM * 2);    // pooled_att (128-row pad)
    float*    pooled = (float*)take((size_t)16 * DIMM * 4);
    _Float16* t16    = (_Float16*)take((size_t)128 * DIMM * 2);    // pool-q LN / head LN (pad)
    // f16 transposed weights (rows padded to mult of 128 — all Ns already are, head->1024)
    _Float16* wt_embed = (_Float16*)take((size_t)DIMM * DIMM * 2);
    _Float16* wt_q[2], *wt_kv[2], *wt_o[2], *wt_1[2], *wt_2[2];
    for (int l = 0; l < 2; l++) {
        wt_q[l]  = (_Float16*)take((size_t)INNER * DIMM * 2);
        wt_kv[l] = (_Float16*)take((size_t)2 * INNER * DIMM * 2);
        wt_o[l]  = (_Float16*)take((size_t)DIMM * INNER * 2);
        wt_1[l]  = (_Float16*)take((size_t)MLPD * DIMM * 2);
        wt_2[l]  = (_Float16*)take((size_t)DIMM * MLPD * 2);
    }
    _Float16* wt_pq  = (_Float16*)take((size_t)INNER * DIMM * 2);
    _Float16* wt_pkv = (_Float16*)take((size_t)2 * INNER * DIMM * 2);
    _Float16* wt_po  = (_Float16*)take((size_t)DIMM * INNER * 2);
    _Float16* wt_head= (_Float16*)take((size_t)1024 * DIMM * 2);

    const size_t ATTN_SMEM = (size_t)(64 * 72) * 2 * 2 + (size_t)64 * 1024 * 2 + 1024 * 4 + 1024;

    // ---- weight conversion/transpose (f32 KxN -> f16 NpadxK) ----
    launch_wtrans(w_embed, wt_embed, DIMM, DIMM, DIMM, stream);
    for (int l = 0; l < 2; l++) {
        launch_wtrans(L[l].wq,  wt_q[l],  DIMM, INNER,     INNER,     stream);
        launch_wtrans(L[l].wkv, wt_kv[l], DIMM, 2 * INNER, 2 * INNER, stream);
        launch_wtrans(L[l].wo,  wt_o[l],  INNER, DIMM,     DIMM,      stream);
        launch_wtrans(L[l].w1,  wt_1[l],  DIMM, MLPD,      MLPD,      stream);
        launch_wtrans(L[l].w2,  wt_2[l],  MLPD, DIMM,      DIMM,      stream);
    }
    launch_wtrans(p_wq,  wt_pq,  DIMM, INNER,     INNER,     stream);
    launch_wtrans(p_wkv, wt_pkv, DIMM, 2 * INNER, 2 * INNER, stream);
    launch_wtrans(p_wo,  wt_po,  INNER, DIMM,     DIMM,      stream);
    launch_wtrans(w_head, wt_head, DIMM, 1000, 1024, stream);

    // ---- embedding ----
    ln_kernel<<<M, 256, 0, stream>>>(patches, pe_ln_g, xn16, DIMM);
    launch_gemm(xn16, wt_embed, b_embed, nullptr, aout16, M, DIMM, DIMM, 0, 0, 1, stream);
    embpos_kernel<<<M, 256, 0, stream>>>(aout16, emb_ln_g, pos_h, pos_w, pos_hw, x);

    // ---- transformer layers ----
    for (int l = 0; l < 2; l++) {
        ln_kernel<<<M, 256, 0, stream>>>(x, L[l].a_norm_g, xn16, DIMM);
        launch_gemm(xn16, wt_q[l],  nullptr, nullptr, q16,  M, INNER,     DIMM, 0, 0, 1, stream);
        launch_gemm(xn16, wt_kv[l], nullptr, nullptr, kv16, M, 2 * INNER, DIMM, 0, 0, 1, stream);
        qk_reorder_kernel<<<dim3(HEADS, NTOK, B), 64, 0, stream>>>(q16,  INNER,     0,     L[l].qn_g, qh, NTOK, 1);
        qk_reorder_kernel<<<dim3(HEADS, NTOK, B), 64, 0, stream>>>(kv16, 2 * INNER, 0,     L[l].kn_g, kh, NTOK, 1);
        qk_reorder_kernel<<<dim3(HEADS, NTOK, B), 64, 0, stream>>>(kv16, 2 * INNER, INNER, nullptr,   vt, NTOK, 2);
        attn_kernel<<<dim3(NTOK / 64, HEADS, B), 128, ATTN_SMEM, stream>>>(qh, kh, vt, image_ids, key_pad, aout16);
        launch_gemm(aout16, wt_o[l], nullptr, x, x, M, DIMM, INNER, 0, 1, 0, stream);
        ln_kernel<<<M, 256, 0, stream>>>(x, L[l].f_norm_g, xn16, DIMM);
        launch_gemm(xn16, wt_1[l], L[l].b1, nullptr, h16, M, MLPD, DIMM, 1, 0, 1, stream);
        launch_gemm(h16,  wt_2[l], L[l].b2, x,       x,   M, DIMM, MLPD, 0, 1, 0, stream);
    }

    // ---- final LN ----
    ln_kernel<<<M, 256, 0, stream>>>(x, final_ln_g, xn16, DIMM);   // xn16 = final-LN(x)

    // ---- attention pooling ----
    launch_gemm(xn16, wt_pkv, nullptr, nullptr, kv16, M, 2 * INNER, DIMM, 0, 0, 1, stream);
    qk_reorder_kernel<<<dim3(HEADS, NTOK, B), 64, 0, stream>>>(kv16, 2 * INNER, 0,     p_kn_g,  kh, NTOK, 1);
    qk_reorder_kernel<<<dim3(HEADS, NTOK, B), 64, 0, stream>>>(kv16, 2 * INNER, INNER, nullptr, vt, NTOK, 2);
    bcast_ln_kernel<<<1, 256, 0, stream>>>(pool_q, p_norm_g, t16);
    launch_gemm(t16, wt_pq, nullptr, nullptr, q16, 16, INNER, DIMM, 0, 0, 1, stream);
    qk_reorder_kernel<<<dim3(HEADS, NI, B), 64, 0, stream>>>(q16, INNER, 0, p_qn_g, qph, NI, 1);
    pool_attn_kernel<<<dim3(NI, HEADS, B), 128, 0, stream>>>(qph, kh, vt, image_ids, key_pad, pat16);
    launch_gemm(pat16, wt_po, nullptr, pool_q, pooled, 16, DIMM, INNER, 0, 2, 0, stream);

    // ---- head ----
    ln_kernel<<<16, 256, 0, stream>>>(pooled, head_ln_g, t16, DIMM);
    launch_gemm(t16, wt_head, nullptr, nullptr, (float*)d_out, 16, 1000, DIMM, 0, 0, 0, stream);
}
